// RingAttention_481036337257
// MI455X (gfx1250) — matrix operand order repaired and checked by reading, not compiled, not run on hardware
//
#include <hip/hip_runtime.h>

#define S_LEN 2048
#define NB    2
#define NH    16
#define HD    128
#define BM    128   // queries per block (8 waves x 16 rows)
#define BN    64    // keys per inner step (4 S-tiles, 32 WMMAs/step)
#define NWAVE 8

typedef _Float16 h16 __attribute__((ext_vector_type(16)));
typedef float    v8f __attribute__((ext_vector_type(8)));

// LDS row strides in halves (padded to dodge bank conflicts)
#define KSTR 136   // 128 + 8
#define VSTR 72    // 64  + 8
#define PSTR 72    // 64  + 8

// ---- 16-lane butterfly reductions via DPP16 (no LDS traffic) ----
template <int CTRL>
__device__ __forceinline__ float dpp16(float v) {
    return __builtin_bit_cast(float,
        __builtin_amdgcn_update_dpp(0, __builtin_bit_cast(int, v),
                                    CTRL, 0xF, 0xF, true));
}
__device__ __forceinline__ float red_max16(float v) {
    v = fmaxf(v, dpp16<0xB1>(v));   // quad_perm [1,0,3,2]  : lane^1
    v = fmaxf(v, dpp16<0x4E>(v));   // quad_perm [2,3,0,1]  : lane^2
    v = fmaxf(v, dpp16<0x141>(v));  // row_half_mirror      : quad^1
    v = fmaxf(v, dpp16<0x140>(v));  // row_mirror           : quad^3
    return v;
}
__device__ __forceinline__ float red_sum16(float v) {
    v += dpp16<0xB1>(v);
    v += dpp16<0x4E>(v);
    v += dpp16<0x141>(v);
    v += dpp16<0x140>(v);
    return v;
}

__global__ __launch_bounds__(256)
void fa_fwd_cdna5(const float* __restrict__ Q, const float* __restrict__ K,
                  const float* __restrict__ V, float* __restrict__ O)
{
    __shared__ _Float16 sK[2][BN * KSTR];      // [key][d], f16
    __shared__ _Float16 sV[2][HD * VSTR];      // transposed: [d][key], f16
    __shared__ _Float16 sP[NWAVE][16 * PSTR];  // per-wave P staging [qrow][key]

    const int tid  = threadIdx.x;
    const int wave = tid >> 5;
    const int lane = tid & 31;
    const int ln   = lane & 15;   // column lane within 16-group
    const int hl   = lane >> 4;   // which half of the wave

    const int m0    = blockIdx.x * BM;
    const int bh    = blockIdx.y;          // b*NH + h
    const int qbase = m0 + wave * 16;

    const size_t row_stride = (size_t)NB * NH * HD;   // 4096 floats between seq positions
    const float* Qb = Q + (size_t)bh * HD;
    const float* Kb = K + (size_t)bh * HD;
    const float* Vb = V + (size_t)bh * HD;

    // 1/sqrt(128) * log2(e): softmax in exp2 domain
    const float qscale = 0.08838834764831845f * 1.4426950408889634f;

    // ---- Q A-fragments: 4 chunks of 16x32 f16 covering D=128 ----
    // A fragment (16x32, contiguous k order used here, same as kb): lane row = lane&15;
// half h (0..15) -> d = d0 + 16*hl + h
    h16 qa[4];
    {
        const float* qrow = Qb + (size_t)(qbase + ln) * row_stride;
        #pragma unroll
        for (int c = 0; c < 4; ++c) {
            const int d0 = 32 * c + 16 * hl;
            #pragma unroll
            for (int i = 0; i < 8; ++i)
                qa[c][i]     = (_Float16)(qrow[d0 + i]      * qscale);
            #pragma unroll
            for (int i = 0; i < 8; ++i)
                qa[c][8 + i] = (_Float16)(qrow[d0 + 8 + i] * qscale);
        }
    }

    v8f acc[8] = {};           // O accumulators: 8 N-tiles of 16 cols, f32
    float mrow[8], lrow[8];    // per-row running max / sum (row = r + 8*hl)
    #pragma unroll
    for (int r = 0; r < 8; ++r) { mrow[r] = -1.0e30f; lrow[r] = 0.0f; }

    // cooperative K/V tile loader: fp32 global -> f16 LDS (V transposed)
    auto loadKV = [&](int k0, int buf) {
        const int r = tid >> 2;          // key row 0..63
        const int c = (tid & 3) * 32;    // 32-wide column chunk
        const float* kp = Kb + (size_t)(k0 + r) * row_stride + c;
        const float* vp = Vb + (size_t)(k0 + r) * row_stride + c;
        _Float16 kh[32], vh[32];
        #pragma unroll
        for (int i = 0; i < 32; ++i) { kh[i] = (_Float16)kp[i]; vh[i] = (_Float16)vp[i]; }
        #pragma unroll
        for (int i = 0; i < 32; ++i) sK[buf][r * KSTR + c + i] = kh[i];
        #pragma unroll
        for (int i = 0; i < 32; ++i) sV[buf][(c + i) * VSTR + r] = vh[i];
    };

    const int nsteps = (m0 + BM) / BN;   // causal: keys in [0, m0+BM)
    loadKV(0, 0);

    for (int step = 0; step < nsteps; ++step) {
        const int k0 = step * BN;
        __syncthreads();
        if (step + 1 < nsteps) loadKV(k0 + BN, (step + 1) & 1);
        const int buf = step & 1;

        // wave-uniform causal skip (EXEC stays all-ones around WMMA)
        if (k0 <= qbase + 15) {
            // ---- S = Q * K^T : four 16x16 f32 tiles over 64 keys ----
            v8f s[4] = {};
            #pragma unroll
            for (int nt = 0; nt < 4; ++nt) {
                #pragma unroll
                for (int c = 0; c < 4; ++c) {
                    // B fragment (32x16, contiguous k order used here, matching qa): col = lane&15; half h -> K = h + 16*hl
                    const int d0 = 32 * c + 16 * hl;
                    h16 kb;
                    #pragma unroll
                    for (int i = 0; i < 16; ++i)
                        kb[i] = sK[buf][(16 * nt + ln) * KSTR + d0 + i];
                    s[nt] = __builtin_amdgcn_wmma_f32_16x16x32_f16(false, qa[c], false, kb,
                                                                   (short)0, s[nt], false, false);
                }
            }

            // ---- causal mask (only diagonal-straddling steps) ----
            if (k0 + BN - 1 > qbase) {
                #pragma unroll
                for (int nt = 0; nt < 4; ++nt) {
                    const int kidx = k0 + 16 * nt + ln;
                    #pragma unroll
                    for (int r = 0; r < 8; ++r) {
                        const int qidx = qbase + r + 8 * hl;
                        if (kidx > qidx) s[nt][r] = -3.0e38f;
                    }
                }
            }

            // ---- online softmax (rows live across 16 lanes; DPP reductions) ----
            #pragma unroll
            for (int r = 0; r < 8; ++r) {
                float t = fmaxf(fmaxf(s[0][r], s[1][r]), fmaxf(s[2][r], s[3][r]));
                t = red_max16(t);
                const float mn    = fmaxf(mrow[r], t);
                const float alpha = exp2f(mrow[r] - mn);
                mrow[r] = mn;
                const float p0 = exp2f(s[0][r] - mn);
                const float p1 = exp2f(s[1][r] - mn);
                const float p2 = exp2f(s[2][r] - mn);
                const float p3 = exp2f(s[3][r] - mn);
                const float rs = red_sum16((p0 + p1) + (p2 + p3));
                lrow[r] = lrow[r] * alpha + rs;
                #pragma unroll
                for (int j = 0; j < 8; ++j) acc[j][r] *= alpha;
                // stash P in C-layout positions (row = r + 8*hl)
                _Float16* pr = &sP[wave][(r + 8 * hl) * PSTR];
                pr[ln]      = (_Float16)p0;
                pr[16 + ln] = (_Float16)p1;
                pr[32 + ln] = (_Float16)p2;
                pr[48 + ln] = (_Float16)p3;
            }

            // ---- reload P as two 16x32 A-fragments (same-wave LDS is in-order) ----
            h16 pa0, pa1;
            #pragma unroll
            for (int i = 0; i < 8; ++i) {
                const _Float16* pr = &sP[wave][ln * PSTR];
                pa0[i]     = pr[16 * hl + i];
                pa0[8 + i] = pr[8 + 16 * hl + i];
                pa1[i]     = pr[32 + 16 * hl + i];
                pa1[8 + i] = pr[40 + 16 * hl + i];
            }

            // ---- O += P * V : V^T staged so B-fragments are contiguous ----
            #pragma unroll
            for (int j = 0; j < 8; ++j) {
                h16 vb0, vb1;  // col = d = 16*j + ln ; half h -> key = kchunk + h + 16*hl
                #pragma unroll
                for (int i = 0; i < 16; ++i) {
                    vb0[i] = sV[buf][(16 * j + ln) * VSTR + 16 * hl + i];
                    vb1[i] = sV[buf][(16 * j + ln) * VSTR + 32 + 16 * hl + i];
                }
                acc[j] = __builtin_amdgcn_wmma_f32_16x16x32_f16(false, pa0, false, vb0,
                                                                (short)0, acc[j], false, false);
                acc[j] = __builtin_amdgcn_wmma_f32_16x16x32_f16(false, pa1, false, vb1,
                                                                (short)0, acc[j], false, false);
            }
        }
    }

    // ---- epilogue: normalize and store, out[b, q, h*HD + d] ----
    const int b = bh / NH;
    const int h = bh % NH;
    #pragma unroll
    for (int r = 0; r < 8; ++r) {
        const float inv = 1.0f / lrow[r];
        const int q = qbase + r + 8 * hl;
        float* orow = O + ((size_t)b * S_LEN + q) * (NH * HD) + (size_t)h * HD;
        #pragma unroll
        for (int j = 0; j < 8; ++j)
            orow[16 * j + ln] = acc[j][r] * inv;
    }
}

extern "C" void kernel_launch(void* const* d_in, const int* in_sizes, int n_in,
                              void* d_out, int out_size, void* d_ws, size_t ws_size,
                              hipStream_t stream) {
    (void)in_sizes; (void)n_in; (void)out_size; (void)d_ws; (void)ws_size;
    const float* Q = (const float*)d_in[0];
    const float* K = (const float*)d_in[1];
    const float* V = (const float*)d_in[2];
    float* O = (float*)d_out;
    dim3 grid(S_LEN / BM, NB * NH);
    fa_fwd_cdna5<<<grid, 256, 0, stream>>>(Q, K, V, O);
}
